// MandoFanbeamFbpLayerNext_19473381720439
// MI455X (gfx1250) — matrix operand order repaired and checked by reading, not compiled, not run on hardware
//
#include <hip/hip_runtime.h>
#include <hip/hip_bf16.h>

// ---------------- problem constants ----------------
#define DET    816
#define VIEWS  720
#define NB     8
#define IMG    512
#define KP     832          // K padded to multiple of 32 for WMMA
#define NROWS  (NB * VIEWS) // 5760 GEMM rows
#define CHUNKS 1632         // (NB*DET*4)/16 : 16-byte chunks per view stage

__device__ __constant__ const float F_DU  = 1.0f;
__device__ __constant__ const float F_OFF = 0.0f;
__device__ __constant__ const float F_SID = 750.0f;
__device__ __constant__ const float F_SDD = 1250.0f;
__device__ __constant__ const float F_PIX = 0.5f;

#define DBETA (6.283185307179586f / (float)VIEWS)

// ---------------- vector types ----------------
typedef __attribute__((ext_vector_type(16))) __bf16 v16bf;
typedef __attribute__((ext_vector_type(8)))  float  v8f;
typedef __attribute__((ext_vector_type(4)))  unsigned int uvec4;
typedef __attribute__((ext_vector_type(4)))  int     v4i;

typedef v4i __attribute__((address_space(1)))* gv4i_p;   // global int4*
typedef v4i __attribute__((address_space(3)))* lv4i_p;   // LDS int4*

// ---------------- async-to-LDS feature probes ----------------
#if __has_builtin(__builtin_amdgcn_global_load_async_to_lds_b128)
#define ASYNC_VIA_BUILTIN 1
#else
#define ASYNC_VIA_BUILTIN 0
#endif

__device__ inline void async_load_16B(const float* g, float* l /*generic ptr into LDS*/) {
#if ASYNC_VIA_BUILTIN
    void* gp = (void*)g;   // drop const
    void* lp = (void*)l;
    __builtin_amdgcn_global_load_async_to_lds_b128((gv4i_p)gp, (lv4i_p)lp, 0, 0);
#else
    // VDST = 32-bit LDS byte address, VADDR = 64-bit global address, no SADDR
    unsigned int laddr = (unsigned int)(unsigned long long)(void*)l;
    unsigned long long gaddr = (unsigned long long)g;
    asm volatile("global_load_async_to_lds_b128 %0, %1, off"
                 :: "v"(laddr), "v"(gaddr) : "memory");
#endif
}

template <int N> __device__ inline void wait_asynccnt() {
#if __has_builtin(__builtin_amdgcn_s_wait_asynccnt)
    __builtin_amdgcn_s_wait_asynccnt((short)N);
#else
    asm volatile("s_wait_asynccnt %0" :: "i"(N) : "memory");
#endif
}

// round-to-nearest-even float -> bf16 (as raw u16)
__device__ inline unsigned short f2bf(float f) {
    unsigned int u = __float_as_uint(f);
    u += 0x7FFFu + ((u >> 16) & 1u);
    return (unsigned short)(u >> 16);
}

// ---------------------------------------------------------------------------
// Kernel 1: build F^T in bf16.  Ft[w][k] = filt[k - w + 407] * cosw[w]*DU*dbeta
// Row-major [DET][KP]; K-pad and out-of-range taps are zero.
// ---------------------------------------------------------------------------
__global__ __launch_bounds__(256) void fbp_build_ft(const float* __restrict__ filt,
                                                    unsigned short* __restrict__ Ft) {
    int idx = blockIdx.x * 256 + threadIdx.x;
    if (idx >= DET * KP) return;
    int w = idx / KP;
    int k = idx - w * KP;
    float val = 0.0f;
    int fi = k - w + 407;                 // center = DET/2 - 1 = 407
    if (k < DET && fi >= 0 && fi < DET) {
        float u    = ((float)w - 407.5f) * F_DU + F_OFF;   // (DET-1)/2 = 407.5
        float cosw = F_SID * rsqrtf(F_SID * F_SID + u * u);
        val = filt[fi] * cosw * F_DU * DBETA;
    }
    Ft[idx] = f2bf(val);
}

// ---------------------------------------------------------------------------
// Kernel 2: convert sinogram to bf16 with K-padding.  Xbf[r][k], r = b*VIEWS+v
// ---------------------------------------------------------------------------
__global__ __launch_bounds__(256) void fbp_conv_x(const float* __restrict__ x,
                                                  unsigned short* __restrict__ Xbf) {
    int idx = blockIdx.x * 256 + threadIdx.x;
    if (idx >= NROWS * KP) return;
    int r = idx / KP;
    int k = idx - r * KP;
    Xbf[idx] = f2bf(k < DET ? x[r * DET + k] : 0.0f);
}

// ---------------------------------------------------------------------------
// Kernel 3: filtering GEMM via v_wmma_f32_16x16x32_bf16.
// Y(5760x816) = Xbf(5760xKP) * Ft^T  (Ft stored as [w][k]).  Each wave computes
// a 16x48 strip (3 N tiles, shared A).  Block = 8 waves = 8 M tiles.
// ---------------------------------------------------------------------------
__global__ __launch_bounds__(256) void fbp_gemm(const unsigned short* __restrict__ Xbf,
                                                const unsigned short* __restrict__ Ft,
                                                float* __restrict__ Y) {
    const int wave = threadIdx.x >> 5;
    const int lane = threadIdx.x & 31;
    const int half = lane >> 4;     // 0: lanes 0-15, 1: lanes 16-31
    const int l16  = lane & 15;

    const int rowBase = (blockIdx.x * 8 + wave) * 16;   // M tile origin
    const int colBase = blockIdx.y * 48;                // N strip origin

    v8f c0 = {}; v8f c1 = {}; v8f c2 = {};

    // A: lane (half,l16) holds row M=l16, K = half*16 .. half*16+15
    const unsigned short* arow = Xbf + (size_t)(rowBase + l16) * KP;

    for (int k = 0; k < KP; k += 32) {
        union { v16bf v; uvec4 u[2]; } A;
        A.u[0] = *reinterpret_cast<const uvec4*>(arow + k + half * 16);
        A.u[1] = *reinterpret_cast<const uvec4*>(arow + k + half * 16 + 8);

        // B: lane (half,l16) holds column N=l16, K = half*16 .. half*16+15
        const unsigned short* b0 = Ft + (size_t)(colBase + 0  + l16) * KP + k + half * 16;
        const unsigned short* b1 = Ft + (size_t)(colBase + 16 + l16) * KP + k + half * 16;
        const unsigned short* b2 = Ft + (size_t)(colBase + 32 + l16) * KP + k + half * 16;

        union { v16bf v; uvec4 u[2]; } B0, B1, B2;
        B0.u[0] = *reinterpret_cast<const uvec4*>(b0);
        B0.u[1] = *reinterpret_cast<const uvec4*>(b0 + 8);
        B1.u[0] = *reinterpret_cast<const uvec4*>(b1);
        B1.u[1] = *reinterpret_cast<const uvec4*>(b1 + 8);
        B2.u[0] = *reinterpret_cast<const uvec4*>(b2);
        B2.u[1] = *reinterpret_cast<const uvec4*>(b2 + 8);

        c0 = __builtin_amdgcn_wmma_f32_16x16x32_bf16(false, A.v, false, B0.v,
                                                     (short)0, c0, false, false);
        c1 = __builtin_amdgcn_wmma_f32_16x16x32_bf16(false, A.v, false, B1.v,
                                                     (short)0, c1, false, false);
        c2 = __builtin_amdgcn_wmma_f32_16x16x32_bf16(false, A.v, false, B2.v,
                                                     (short)0, c2, false, false);
    }

    // C/D layout: VGPR r -> M = r + 8*half, N = l16
    #pragma unroll
    for (int r = 0; r < 8; ++r) {
        size_t row = (size_t)(rowBase + half * 8 + r) * DET;
        Y[row + colBase + 0  + l16] = c0[r];
        Y[row + colBase + 16 + l16] = c1[r];
        Y[row + colBase + 32 + l16] = c2[r];
    }
}

// ---------------------------------------------------------------------------
// Kernel 4: fan-beam backprojection with double-buffered async LDS staging.
// One block = 16x16 pixel tile, all 8 batches (8 accumulators/thread).
// While computing view v from buffer v&1, view v+1 streams into the other
// buffer via global_load_async_to_lds_b128; s_wait_asynccnt retires exactly
// the previous batch of 7 per-wave async issues (in-order completion).
// ---------------------------------------------------------------------------
__device__ inline void stage_view(const float* __restrict__ sino, int v, float* lbuf) {
    #pragma unroll
    for (int j = 0; j < 7; ++j) {
        int c = threadIdx.x + 256 * j;
        c = c < (CHUNKS - 1) ? c : (CHUNKS - 1);  // clamp: dup writes are benign
        int b = c / 204;                          // 204 = DET*4/16 chunks per row
        int d = c - b * 204;                      // 16B chunk within row
        const float* g = sino + (size_t)(b * VIEWS + v) * DET + d * 4;
        async_load_16B(g, lbuf + c * 4);
    }
}

__global__ __launch_bounds__(256) void fbp_backproject(const float* __restrict__ sino,
                                                       float* __restrict__ out) {
    __shared__ __align__(16) float rows[2][NB * DET];   // 2 x 26112 B

    const int tx = threadIdx.x & 15;
    const int ty = threadIdx.x >> 4;
    const int px = blockIdx.x * 16 + tx;
    const int py = blockIdx.y * 16 + ty;

    const float Xc = ((float)px - 255.5f) * F_PIX;   // (IMG-1)/2 = 255.5
    const float Yc = -(((float)py - 255.5f) * F_PIX);

    float acc[NB];
    #pragma unroll
    for (int b = 0; b < NB; ++b) acc[b] = 0.0f;

    stage_view(sino, 0, rows[0]);                 // prologue: view 0 -> buf 0

    #pragma unroll 1
    for (int v = 0; v < VIEWS; ++v) {
        const int cur = v & 1;
        float* cbuf = rows[cur];

        __syncthreads();                          // readers of rows[cur^1] done
        if (v + 1 < VIEWS) {
            stage_view(sino, v + 1, rows[cur ^ 1]);
            wait_asynccnt<7>();                   // cur's 7 done; next 7 in flight
        } else {
            wait_asynccnt<0>();                   // drain final batch
        }
        __syncthreads();                          // all waves' cur data resident

        float beta = (float)v * DBETA;
        float sb, cb;
        __sincosf(beta, &sb, &cb);

        float xr   = Xc * cb + Yc * sb;
        float yr   = -Xc * sb + Yc * cb;
        float L    = F_SID - yr;
        float invL = 1.0f / L;
        float u    = F_SDD * xr * invL;
        float w    = (F_SID * F_SID) * invL * invL;
        float t    = (u - F_OFF) * (1.0f / F_DU) + 407.5f;

        int i0 = (int)floorf(t);
        i0 = i0 < 0 ? 0 : (i0 > DET - 2 ? DET - 2 : i0);
        float frac = t - (float)i0;
        float wi = (t >= 0.0f && t <= (float)(DET - 1)) ? w : 0.0f;

        #pragma unroll
        for (int b = 0; b < NB; ++b) {
            float v0 = cbuf[b * DET + i0];
            float v1 = cbuf[b * DET + i0 + 1];
            acc[b] += wi * (v0 + frac * (v1 - v0));
        }
    }

    #pragma unroll
    for (int b = 0; b < NB; ++b)
        out[(size_t)b * IMG * IMG + (size_t)py * IMG + px] = acc[b];
}

// ---------------------------------------------------------------------------
extern "C" void kernel_launch(void* const* d_in, const int* in_sizes, int n_in,
                              void* d_out, int out_size, void* d_ws, size_t ws_size,
                              hipStream_t stream) {
    const float* x    = (const float*)d_in[0];   // (8,720,816) f32
    const float* filt = (const float*)d_in[1];   // (816,) f32
    float* out        = (float*)d_out;           // (8,512,512) f32

    char* ws = (char*)d_ws;
    size_t off = 0;
    unsigned short* Ft  = (unsigned short*)(ws + off);
    off += ((size_t)DET * KP * 2 + 255) & ~(size_t)255;
    unsigned short* Xbf = (unsigned short*)(ws + off);
    off += ((size_t)NROWS * KP * 2 + 255) & ~(size_t)255;
    float* Ysino        = (float*)(ws + off);
    // total ws use ~29.8 MB

    {   // build weighted Toeplitz filter matrix (bf16)
        int n = DET * KP;
        fbp_build_ft<<<(n + 255) / 256, 256, 0, stream>>>(filt, Ft);
    }
    {   // sinogram -> bf16, K-padded
        int n = NROWS * KP;
        fbp_conv_x<<<(n + 255) / 256, 256, 0, stream>>>(x, Xbf);
    }
    {   // filtering GEMM: 360 M tiles / 8 per block, 51 N tiles / 3 per wave
        dim3 grid(45, 17);
        fbp_gemm<<<grid, 256, 0, stream>>>(Xbf, Ft, Ysino);
    }
    {   // backprojection over 16x16 pixel tiles
        dim3 grid(IMG / 16, IMG / 16);
        fbp_backproject<<<grid, 256, 0, stream>>>(Ysino, out);
    }
}
